// NeuronMiniMaxM2DecoderLayerV3_65189013619208
// MI455X (gfx1250) — compile-verified, matrix operand order checked by reading, not run on hardware
//
#include <hip/hip_runtime.h>
#include <hip/hip_bf16.h>
#include <math.h>

// ---------------- problem constants ----------------
#define DDIM   2048
#define SEQ    2048
#define NH     16
#define NKV    4
#define DHEAD  128
#define RDIM   64
#define NEXP   8
#define IDIM   1024
#define EPSV   1e-6f
#define THETAV 1000000.0f

typedef __attribute__((ext_vector_type(16))) __bf16 v16bf;
typedef __attribute__((ext_vector_type(8)))  float  v8f;

union FragB { uint4 u4[2]; v16bf v; };   // 32 bytes = one 16x32 bf16 A/B fragment per lane

static __device__ __forceinline__ unsigned short f2bfu(float f) {
    union { __hip_bfloat16 h; unsigned short u; } cv;
    cv.h = __float2bfloat16(f);
    return cv.u;
}

// 32-bit LDS byte offset of a __shared__ object (generic -> AS3 -> int)
static __device__ __forceinline__ unsigned lds_addr_of(const void* p) {
    return (unsigned)(__SIZE_TYPE__)(__attribute__((address_space(3))) const char*)p;
}

// ---------------- elementwise f32 -> bf16 cast ----------------
__global__ void cast_bf16_kernel(const float* __restrict__ in,
                                 __hip_bfloat16* __restrict__ out, long long n) {
    long long i = (long long)blockIdx.x * blockDim.x + threadIdx.x;
    long long stride = (long long)gridDim.x * blockDim.x;
    for (; i < n; i += stride) out[i] = __float2bfloat16(in[i]);
}

// ---------------- (a [+ b]) -> x1, rmsnorm -> bf16 (+ f32) ----------------
__global__ __launch_bounds__(256) void add_rmsnorm_kernel(
    const float* __restrict__ a, const float* __restrict__ b,
    const float* __restrict__ w,
    float* __restrict__ sum_out,           // may be null
    __hip_bfloat16* __restrict__ bf_out,   // normed, bf16
    float* __restrict__ f_out)             // normed, f32 (may be null)
{
    __shared__ float row[DDIM];
    __shared__ float red[256];
    int t = blockIdx.x, tid = threadIdx.x;
    const float* ap = a + (long long)t * DDIM;
    const float* bp = b ? (b + (long long)t * DDIM) : nullptr;
    float ss = 0.f;
    for (int d = tid; d < DDIM; d += 256) {
        float v = ap[d] + (bp ? bp[d] : 0.f);
        row[d] = v;
        ss += v * v;
    }
    red[tid] = ss; __syncthreads();
    for (int s = 128; s > 0; s >>= 1) { if (tid < s) red[tid] += red[tid + s]; __syncthreads(); }
    float rinv = rsqrtf(red[0] / (float)DDIM + EPSV);
    for (int d = tid; d < DDIM; d += 256) {
        float v = row[d];
        if (sum_out) sum_out[(long long)t * DDIM + d] = v;
        float nv = v * rinv * w[d];
        bf_out[(long long)t * DDIM + d] = __float2bfloat16(nv);
        if (f_out) f_out[(long long)t * DDIM + d] = nv;
    }
}

// ---------------- tiled bf16 GEMM: C(f32 MxN) = A(bf16 MxK) x B(bf16 KxN) ----------------
// block = 256 threads = 8 waves; macro tile 128x128, K-step 32, double-buffered LDS.
// A tile staged with global_load_async_to_lds_b128 (ASYNCcnt path, no VGPR round trip).
// B tile staged through VGPRs with (k,k+1) bf16 pairs packed into dwords -> ds_store_b32,
// which directly matches the 16-bit WMMA B-fragment dword layout.
#define BM 128
#define BN 128
#define BK 32

__global__ __launch_bounds__(256) void gemm_bf16_kernel(
    const __hip_bfloat16* __restrict__ A,
    const __hip_bfloat16* __restrict__ B,
    float* __restrict__ C,
    int M, int N, int K, int beta)
{
    __shared__ unsigned short As[2][BM * BK];          // [buf][row][k]      8KB each
    __shared__ unsigned int   Bs[2][BN * (BK / 2)];    // [buf][n][kpair]    8KB each
    int tid  = threadIdx.x;
    int wave = tid >> 5;
    int lane = tid & 31;
    int waveM = wave >> 1;       // 0..3
    int waveN = wave & 1;        // 0..1
    int bm0 = blockIdx.y * BM;
    int bn0 = blockIdx.x * BN;
    int l15 = lane & 15;
    int hb  = lane >> 4;
    int kb  = hb ? 8 : 0;        // per-lane K base of the 16-bit fragment layout

    // staging roles
    int arow = tid >> 1;             // A: row 0..127
    int ac0  = (tid & 1) * 16;       // A: element 0 or 16 (2 x b128 per thread)
    int bkp  = tid >> 4;             // B: k-pair 0..15
    int bnt  = (tid & 15) * 8;       // B: 8 consecutive n

    v8f acc[2][4];
    for (int i = 0; i < 2; ++i)
        for (int j = 0; j < 4; ++j)
            for (int r = 0; r < 8; ++r) acc[i][j][r] = 0.f;

    uint4 br0, br1;   // B staging registers

    int nk = K / BK;

    // ---- prologue: stage tile 0 ----
    {
        const __hip_bfloat16* g = A + (long long)(bm0 + arow) * K + ac0;
        unsigned lds = lds_addr_of(&As[0][arow * BK + ac0]);
        asm volatile("global_load_async_to_lds_b128 %0, %1, off"
                     :: "v"(lds), "v"(g) : "memory");
        asm volatile("global_load_async_to_lds_b128 %0, %1, off offset:16"
                     :: "v"(lds), "v"(g) : "memory");
        const __hip_bfloat16* bsrc = B + (long long)(2 * bkp) * N + bn0 + bnt;
        br0 = *reinterpret_cast<const uint4*>(bsrc);
        br1 = *reinterpret_cast<const uint4*>(bsrc + N);
        unsigned short t0[8], t1[8];
        *reinterpret_cast<uint4*>(t0) = br0;
        *reinterpret_cast<uint4*>(t1) = br1;
        for (int i = 0; i < 8; ++i)
            Bs[0][(bnt + i) * (BK / 2) + bkp] = (unsigned)t0[i] | ((unsigned)t1[i] << 16);
    }
    asm volatile("s_wait_asynccnt 0x0" ::: "memory");
    __syncthreads();

    for (int kc = 0; kc < nk; ++kc) {
        int cur = kc & 1, nxt = cur ^ 1;
        bool more = (kc + 1 < nk);

        // ---- kick off next tile: A async copy + B global loads to regs ----
        if (more) {
            const __hip_bfloat16* g = A + (long long)(bm0 + arow) * K + (kc + 1) * BK + ac0;
            unsigned lds = lds_addr_of(&As[nxt][arow * BK + ac0]);
            asm volatile("global_load_async_to_lds_b128 %0, %1, off"
                         :: "v"(lds), "v"(g) : "memory");
            asm volatile("global_load_async_to_lds_b128 %0, %1, off offset:16"
                         :: "v"(lds), "v"(g) : "memory");
            const __hip_bfloat16* bsrc = B + (long long)((kc + 1) * BK + 2 * bkp) * N + bn0 + bnt;
            br0 = *reinterpret_cast<const uint4*>(bsrc);
            br1 = *reinterpret_cast<const uint4*>(bsrc + N);
        }

        // ---- WMMA on current buffers ----
        const unsigned short* as   = As[cur];
        const unsigned short* bs16 = reinterpret_cast<const unsigned short*>(Bs[cur]);
        FragB afr[2], bfr[4];
        for (int mt = 0; mt < 2; ++mt) {
            int row = waveM * 32 + mt * 16 + l15;
            afr[mt].u4[0] = *reinterpret_cast<const uint4*>(&as[row * BK + kb]);
            afr[mt].u4[1] = *reinterpret_cast<const uint4*>(&as[row * BK + kb + 16]);
        }
        for (int nt = 0; nt < 4; ++nt) {
            int n = waveN * 64 + nt * 16 + l15;
            bfr[nt].u4[0] = *reinterpret_cast<const uint4*>(&bs16[n * BK + kb]);
            bfr[nt].u4[1] = *reinterpret_cast<const uint4*>(&bs16[n * BK + kb + 16]);
        }
        for (int mt = 0; mt < 2; ++mt)
            for (int nt = 0; nt < 4; ++nt)
                acc[mt][nt] = __builtin_amdgcn_wmma_f32_16x16x32_bf16(
                    false, afr[mt].v, false, bfr[nt].v, (short)0, acc[mt][nt], false, false);

        // ---- finish staging next tile ----
        if (more) {
            unsigned short t0[8], t1[8];
            *reinterpret_cast<uint4*>(t0) = br0;
            *reinterpret_cast<uint4*>(t1) = br1;
            for (int i = 0; i < 8; ++i)
                Bs[nxt][(bnt + i) * (BK / 2) + bkp] = (unsigned)t0[i] | ((unsigned)t1[i] << 16);
        }
        asm volatile("s_wait_asynccnt 0x0" ::: "memory");
        __syncthreads();
    }

    // ---- epilogue: C layout rows m = r + 8*(lane>>4), col = lane&15 ----
    for (int mt = 0; mt < 2; ++mt)
        for (int nt = 0; nt < 4; ++nt)
            for (int r = 0; r < 8; ++r) {
                int row = bm0 + waveM * 32 + mt * 16 + r + hb * 8;
                int col = bn0 + waveN * 64 + nt * 16 + l15;
                long long idx = (long long)row * N + col;
                float v = acc[mt][nt][r];
                if (beta) v += C[idx];
                C[idx] = v;
            }
}

// ---------------- qk-norm + RoPE + layout change + bf16 cast ----------------
// q,k,v: [S][HD] f32 rows. Outputs:
//   qh [NH][S][DHEAD] bf16 (scaled by 1/sqrt(DHEAD)), kh [NKV][S][DHEAD] bf16,
//   vT [NKV][DHEAD][S] bf16 (transposed for P*V WMMA B-fragments)
__global__ __launch_bounds__(256) void qkrope_kernel(
    const float* __restrict__ q, const float* __restrict__ k, const float* __restrict__ v,
    const float* __restrict__ qw, const float* __restrict__ kw,
    __hip_bfloat16* __restrict__ qh, __hip_bfloat16* __restrict__ kh,
    __hip_bfloat16* __restrict__ vT)
{
    __shared__ float qrow[NH * DHEAD];
    __shared__ float krow[NKV * DHEAD];
    __shared__ float red[256];
    int s = blockIdx.x, tid = threadIdx.x;
    const float* qp = q + (long long)s * (NH * DHEAD);
    const float* kp = k + (long long)s * (NKV * DHEAD);
    float ssq = 0.f, ssk = 0.f;
    for (int d = tid; d < NH * DHEAD; d += 256)  { float x = qp[d]; qrow[d] = x; ssq += x * x; }
    for (int d = tid; d < NKV * DHEAD; d += 256) { float x = kp[d]; krow[d] = x; ssk += x * x; }
    red[tid] = ssq; __syncthreads();
    for (int t = 128; t > 0; t >>= 1) { if (tid < t) red[tid] += red[tid + t]; __syncthreads(); }
    float qr = rsqrtf(red[0] / (float)(NH * DHEAD) + EPSV);
    __syncthreads();
    red[tid] = ssk; __syncthreads();
    for (int t = 128; t > 0; t >>= 1) { if (tid < t) red[tid] += red[tid + t]; __syncthreads(); }
    float kr = rsqrtf(red[0] / (float)(NKV * DHEAD) + EPSV);

    const float SC = 0.08838834764831845f;  // 1/sqrt(128), folded into q
    const int HRD = RDIM / 2;               // 32 rotated pairs per head

    // q: rotated pairs
    for (int p = tid; p < NH * HRD; p += 256) {
        int h = p / HRD, i = p % HRD;
        int base = h * DHEAD;
        float v0 = qrow[base + i]       * qr * qw[base + i];
        float v1 = qrow[base + i + HRD] * qr * qw[base + i + HRD];
        float ang = (float)s * __powf(THETAV, -(float)i / (float)HRD);
        float c = __cosf(ang), sn = __sinf(ang);
        long long o = ((long long)h * SEQ + s) * DHEAD;
        qh[o + i]       = __float2bfloat16((v0 * c - v1 * sn) * SC);
        qh[o + i + HRD] = __float2bfloat16((v1 * c + v0 * sn) * SC);
    }
    // q: pass-through dims [RDIM, DHEAD)
    for (int p = tid; p < NH * (DHEAD - RDIM); p += 256) {
        int h = p / (DHEAD - RDIM), d = RDIM + p % (DHEAD - RDIM);
        long long o = ((long long)h * SEQ + s) * DHEAD;
        qh[o + d] = __float2bfloat16(qrow[h * DHEAD + d] * qr * qw[h * DHEAD + d] * SC);
    }
    // k: rotated pairs
    for (int p = tid; p < NKV * HRD; p += 256) {
        int h = p / HRD, i = p % HRD;
        int base = h * DHEAD;
        float v0 = krow[base + i]       * kr * kw[base + i];
        float v1 = krow[base + i + HRD] * kr * kw[base + i + HRD];
        float ang = (float)s * __powf(THETAV, -(float)i / (float)HRD);
        float c = __cosf(ang), sn = __sinf(ang);
        long long o = ((long long)h * SEQ + s) * DHEAD;
        kh[o + i]       = __float2bfloat16(v0 * c - v1 * sn);
        kh[o + i + HRD] = __float2bfloat16(v1 * c + v0 * sn);
    }
    // k: pass-through
    for (int p = tid; p < NKV * (DHEAD - RDIM); p += 256) {
        int h = p / (DHEAD - RDIM), d = RDIM + p % (DHEAD - RDIM);
        long long o = ((long long)h * SEQ + s) * DHEAD;
        kh[o + d] = __float2bfloat16(krow[h * DHEAD + d] * kr * kw[h * DHEAD + d]);
    }
    // v: transpose to [kv][dh][s]
    for (int p = tid; p < NKV * DHEAD; p += 256) {
        int g = p / DHEAD, d = p % DHEAD;
        vT[((long long)g * DHEAD + d) * SEQ + s] =
            __float2bfloat16(v[(long long)s * (NKV * DHEAD) + g * DHEAD + d]);
    }
}

// ---------------- flash attention: one wave per (16-query tile, head) ----------------
__global__ __launch_bounds__(32) void attn_kernel(
    const __hip_bfloat16* __restrict__ qh,   // [NH][S][DHEAD], pre-scaled by 1/sqrt(DHEAD)
    const __hip_bfloat16* __restrict__ kh,   // [NKV][S][DHEAD]
    const __hip_bfloat16* __restrict__ vT,   // [NKV][DHEAD][S]
    __hip_bfloat16* __restrict__ obf)        // [S][NH*DHEAD]
{
    __shared__ unsigned short Ps[16 * 32];   // softmax tile, A-fragment staging
    int qb = blockIdx.x, h = blockIdx.y;
    int lane = threadIdx.x;
    int kvh = h / (NH / NKV);
    int l15 = lane & 15;
    int hb  = lane >> 4;
    int kb  = hb ? 8 : 0;

    // preload Q A-fragments: 4 dh-chunks of 32
    FragB qf[4];
    {
        const __hip_bfloat16* qp = qh + ((long long)h * SEQ + qb * 16 + l15) * DHEAD;
        for (int c = 0; c < 4; ++c) {
            qf[c].u4[0] = *reinterpret_cast<const uint4*>(qp + c * 32 + kb);
            qf[c].u4[1] = *reinterpret_cast<const uint4*>(qp + c * 32 + kb + 16);
        }
    }
    v8f oacc[8];
    for (int t = 0; t < 8; ++t) for (int r = 0; r < 8; ++r) oacc[t][r] = 0.f;
    float mrow[8], lrow[8];
    for (int r = 0; r < 8; ++r) { mrow[r] = -3.0e38f; lrow[r] = 0.f; }

    int nchunk = (qb * 16 + 15) / 32 + 1;    // causal key coverage, 32 keys per chunk
    for (int kc = 0; kc < nchunk; ++kc) {
        // scores: S = Q x K^T   (two 16-key groups)
        v8f sacc[2];
        for (int g = 0; g < 2; ++g) {
            v8f sc; for (int r = 0; r < 8; ++r) sc[r] = 0.f;
            int key = kc * 32 + g * 16 + l15;
            const __hip_bfloat16* kp = kh + ((long long)kvh * SEQ + key) * DHEAD;
            for (int c = 0; c < 4; ++c) {
                FragB kf;
                kf.u4[0] = *reinterpret_cast<const uint4*>(kp + c * 32 + kb);
                kf.u4[1] = *reinterpret_cast<const uint4*>(kp + c * 32 + kb + 16);
                sc = __builtin_amdgcn_wmma_f32_16x16x32_bf16(
                    false, qf[c].v, false, kf.v, (short)0, sc, false, false);
            }
            sacc[g] = sc;
        }
        // causal mask + online softmax (row stats via width-16 shuffles, wave32)
        float p0a[8], p1a[8];
        for (int r = 0; r < 8; ++r) {
            int qrow = qb * 16 + r + hb * 8;
            int key0 = kc * 32 + l15;
            float s0 = (key0      <= qrow) ? sacc[0][r] : -3.0e38f;
            float s1 = (key0 + 16 <= qrow) ? sacc[1][r] : -3.0e38f;
            float mx = fmaxf(s0, s1);
            for (int off = 1; off < 16; off <<= 1) mx = fmaxf(mx, __shfl_xor(mx, off, 16));
            float mnew  = fmaxf(mrow[r], mx);
            float scale = __expf(mrow[r] - mnew);
            float p0 = __expf(s0 - mnew);
            float p1 = __expf(s1 - mnew);
            float rs = p0 + p1;
            for (int off = 1; off < 16; off <<= 1) rs += __shfl_xor(rs, off, 16);
            lrow[r] = lrow[r] * scale + rs;
            mrow[r] = mnew;
            for (int t = 0; t < 8; ++t) oacc[t][r] *= scale;
            p0a[r] = p0; p1a[r] = p1;
        }
        // relayout P (C-layout -> A-fragment layout) through LDS
        for (int r = 0; r < 8; ++r) {
            int m = r + hb * 8;
            Ps[m * 32 + l15]      = f2bfu(p0a[r]);
            Ps[m * 32 + 16 + l15] = f2bfu(p1a[r]);
        }
        __syncthreads();
        FragB pf;
        pf.u4[0] = *reinterpret_cast<const uint4*>(&Ps[l15 * 32 + kb]);
        pf.u4[1] = *reinterpret_cast<const uint4*>(&Ps[l15 * 32 + kb + 16]);
        // O += P x V  (8 output dh groups of 16)
        for (int t = 0; t < 8; ++t) {
            FragB vf;
            const __hip_bfloat16* vp = vT + ((long long)kvh * DHEAD + t * 16 + l15) * SEQ + kc * 32;
            vf.u4[0] = *reinterpret_cast<const uint4*>(vp + kb);
            vf.u4[1] = *reinterpret_cast<const uint4*>(vp + kb + 16);
            oacc[t] = __builtin_amdgcn_wmma_f32_16x16x32_bf16(
                false, pf.v, false, vf.v, (short)0, oacc[t], false, false);
        }
        __syncthreads();
    }
    // normalize + write
    for (int r = 0; r < 8; ++r) {
        float inv = 1.f / lrow[r];
        int qrow = qb * 16 + r + hb * 8;
        for (int t = 0; t < 8; ++t) {
            int col = h * DHEAD + t * 16 + l15;
            obf[(long long)qrow * (NH * DHEAD) + col] = __float2bfloat16(oacc[t][r] * inv);
        }
    }
}

// ---------------- router: sigmoid + bias top-2, normalized affinities ----------------
__global__ __launch_bounds__(256) void router_kernel(
    const float* __restrict__ h2, const float* __restrict__ rw, const float* __restrict__ rb,
    int* __restrict__ topi, float* __restrict__ aff)
{
    __shared__ float red[NEXP * 256];
    int t = blockIdx.x, tid = threadIdx.x;
    float part[NEXP];
    for (int e = 0; e < NEXP; ++e) part[e] = 0.f;
    const float* hp = h2 + (long long)t * DDIM;
    for (int d = tid; d < DDIM; d += 256) {
        float x = hp[d];
        const float* wr = rw + (long long)d * NEXP;
        for (int e = 0; e < NEXP; ++e) part[e] += x * wr[e];
    }
    for (int e = 0; e < NEXP; ++e) red[e * 256 + tid] = part[e];
    __syncthreads();
    for (int s = 128; s > 0; s >>= 1) {
        if (tid < s) for (int e = 0; e < NEXP; ++e) red[e * 256 + tid] += red[e * 256 + tid + s];
        __syncthreads();
    }
    if (tid == 0) {
        float sg[NEXP], score[NEXP];
        for (int e = 0; e < NEXP; ++e) {
            sg[e] = 1.f / (1.f + __expf(-red[e * 256]));
            score[e] = sg[e] + rb[e];
        }
        int i0 = 0;
        for (int e = 1; e < NEXP; ++e) if (score[e] > score[i0]) i0 = e;
        int i1 = (i0 == 0) ? 1 : 0;
        for (int e = 0; e < NEXP; ++e) { if (e == i0) continue; if (score[e] > score[i1]) i1 = e; }
        float a0 = sg[i0], a1 = sg[i1], s = a0 + a1;
        topi[t * 2] = i0; topi[t * 2 + 1] = i1;
        aff[t * 2] = a0 / s; aff[t * 2 + 1] = a1 / s;
    }
}

// ---------------- act = silu(gate) * up * combine[t][e]  -> bf16 ----------------
__global__ void act_kernel(const float* __restrict__ gate, const float* __restrict__ up,
                           const int* __restrict__ topi, const float* __restrict__ aff,
                           int e, __hip_bfloat16* __restrict__ act)
{
    long long n = (long long)SEQ * IDIM;
    long long i = (long long)blockIdx.x * blockDim.x + threadIdx.x;
    long long stride = (long long)gridDim.x * blockDim.x;
    for (; i < n; i += stride) {
        int t = (int)(i / IDIM);
        float sc = (topi[t * 2] == e) ? aff[t * 2]
                 : (topi[t * 2 + 1] == e) ? aff[t * 2 + 1] : 0.f;
        float g = gate[i];
        float silu = g / (1.f + __expf(-g));
        act[i] = __float2bfloat16(silu * up[i] * sc);
    }
}

// ---------------- out = a + b (f32) ----------------
__global__ void add_kernel(const float* __restrict__ a, const float* __restrict__ b,
                           float* __restrict__ out, long long n) {
    long long i = (long long)blockIdx.x * blockDim.x + threadIdx.x;
    long long stride = (long long)gridDim.x * blockDim.x;
    for (; i < n; i += stride) out[i] = a[i] + b[i];
}

// ================= host-side orchestration =================
extern "C" void kernel_launch(void* const* d_in, const int* in_sizes, int n_in,
                              void* d_out, int out_size, void* d_ws, size_t ws_size,
                              hipStream_t stream) {
    (void)in_sizes; (void)n_in; (void)out_size; (void)ws_size;
    const float* x     = (const float*)d_in[0];
    const float* ln1_w = (const float*)d_in[1];
    const float* ln2_w = (const float*)d_in[2];
    const float* wq    = (const float*)d_in[3];
    const float* wk    = (const float*)d_in[4];
    const float* wv    = (const float*)d_in[5];
    const float* wo    = (const float*)d_in[6];
    const float* qnw   = (const float*)d_in[7];
    const float* knw   = (const float*)d_in[8];
    const float* rw    = (const float*)d_in[9];
    const float* rb    = (const float*)d_in[10];
    const float* wg    = (const float*)d_in[11];
    const float* wu    = (const float*)d_in[12];
    const float* wd    = (const float*)d_in[13];
    float* out = (float*)d_out;

    char* ws = (char*)d_ws;
    size_t off = 0;
    auto walloc = [&](size_t bytes) -> void* {
        void* p = ws + off;
        off = (off + bytes + 255) & ~(size_t)255;
        return p;
    };
    const long long SD  = (long long)SEQ * DDIM;            // 4M
    const long long KVD = (long long)SEQ * NKV * DHEAD;     // 1M
    const long long EDI = (long long)NEXP * DDIM * IDIM;    // 16.8M

    __hip_bfloat16* wq_b = (__hip_bfloat16*)walloc((long long)DDIM * (NH * DHEAD) * 2);
    __hip_bfloat16* wk_b = (__hip_bfloat16*)walloc((long long)DDIM * (NKV * DHEAD) * 2);
    __hip_bfloat16* wv_b = (__hip_bfloat16*)walloc((long long)DDIM * (NKV * DHEAD) * 2);
    __hip_bfloat16* wo_b = (__hip_bfloat16*)walloc((long long)(NH * DHEAD) * DDIM * 2);
    __hip_bfloat16* wg_b = (__hip_bfloat16*)walloc(EDI * 2);
    __hip_bfloat16* wu_b = (__hip_bfloat16*)walloc(EDI * 2);
    __hip_bfloat16* wd_b = (__hip_bfloat16*)walloc(EDI * 2);
    __hip_bfloat16* h_b  = (__hip_bfloat16*)walloc(SD * 2);
    float* q_f  = (float*)walloc(SD * 4);
    float* k_f  = (float*)walloc(KVD * 4);
    float* v_f  = (float*)walloc(KVD * 4);
    __hip_bfloat16* qh_b = (__hip_bfloat16*)walloc(SD * 2);
    __hip_bfloat16* kh_b = (__hip_bfloat16*)walloc(KVD * 2);
    __hip_bfloat16* vT_b = (__hip_bfloat16*)walloc(KVD * 2);
    __hip_bfloat16* o_b  = (__hip_bfloat16*)walloc(SD * 2);
    float* attn_f = (float*)walloc(SD * 4);
    float* x1_f   = (float*)walloc(SD * 4);
    __hip_bfloat16* h2_b = (__hip_bfloat16*)walloc(SD * 2);
    float* h2_f  = (float*)walloc(SD * 4);
    int*   topi  = (int*)walloc((long long)SEQ * 2 * 4);
    float* aff   = (float*)walloc((long long)SEQ * 2 * 4);
    float* gate_f = (float*)walloc((long long)SEQ * IDIM * 4);
    float* up_f   = (float*)walloc((long long)SEQ * IDIM * 4);
    __hip_bfloat16* act_b = (__hip_bfloat16*)walloc((long long)SEQ * IDIM * 2);
    float* moe_f = (float*)walloc(SD * 4);

    // 1) weight casts to bf16
    cast_bf16_kernel<<<2048, 256, 0, stream>>>(wq, wq_b, (long long)DDIM * NH * DHEAD);
    cast_bf16_kernel<<<1024, 256, 0, stream>>>(wk, wk_b, (long long)DDIM * NKV * DHEAD);
    cast_bf16_kernel<<<1024, 256, 0, stream>>>(wv, wv_b, (long long)DDIM * NKV * DHEAD);
    cast_bf16_kernel<<<2048, 256, 0, stream>>>(wo, wo_b, (long long)NH * DHEAD * DDIM);
    cast_bf16_kernel<<<4096, 256, 0, stream>>>(wg, wg_b, EDI);
    cast_bf16_kernel<<<4096, 256, 0, stream>>>(wu, wu_b, EDI);
    cast_bf16_kernel<<<4096, 256, 0, stream>>>(wd, wd_b, EDI);

    // 2) ln1: h = rmsnorm(x) -> bf16
    add_rmsnorm_kernel<<<SEQ, 256, 0, stream>>>(x, nullptr, ln1_w, nullptr, h_b, nullptr);

    // 3) QKV GEMMs
    gemm_bf16_kernel<<<dim3((NH * DHEAD) / BN, SEQ / BM), 256, 0, stream>>>(h_b, wq_b, q_f, SEQ, NH * DHEAD, DDIM, 0);
    gemm_bf16_kernel<<<dim3((NKV * DHEAD) / BN, SEQ / BM), 256, 0, stream>>>(h_b, wk_b, k_f, SEQ, NKV * DHEAD, DDIM, 0);
    gemm_bf16_kernel<<<dim3((NKV * DHEAD) / BN, SEQ / BM), 256, 0, stream>>>(h_b, wv_b, v_f, SEQ, NKV * DHEAD, DDIM, 0);

    // 4) qk-norm + RoPE + head layouts
    qkrope_kernel<<<SEQ, 256, 0, stream>>>(q_f, k_f, v_f, qnw, knw, qh_b, kh_b, vT_b);

    // 5) flash attention
    attn_kernel<<<dim3(SEQ / 16, NH), 32, 0, stream>>>(qh_b, kh_b, vT_b, o_b);

    // 6) output projection
    gemm_bf16_kernel<<<dim3(DDIM / BN, SEQ / BM), 256, 0, stream>>>(o_b, wo_b, attn_f, SEQ, DDIM, NH * DHEAD, 0);

    // 7) residual + ln2
    add_rmsnorm_kernel<<<SEQ, 256, 0, stream>>>(x, attn_f, ln2_w, x1_f, h2_b, h2_f);

    // 8) router
    router_kernel<<<SEQ, 256, 0, stream>>>(h2_f, rw, rb, topi, aff);

    // 9) MoE: dense over experts; down-proj accumulates into moe_f
    for (int e = 0; e < NEXP; ++e) {
        const __hip_bfloat16* wge = wg_b + (long long)e * DDIM * IDIM;
        const __hip_bfloat16* wue = wu_b + (long long)e * DDIM * IDIM;
        const __hip_bfloat16* wde = wd_b + (long long)e * IDIM * DDIM;
        gemm_bf16_kernel<<<dim3(IDIM / BN, SEQ / BM), 256, 0, stream>>>(h2_b, wge, gate_f, SEQ, IDIM, DDIM, 0);
        gemm_bf16_kernel<<<dim3(IDIM / BN, SEQ / BM), 256, 0, stream>>>(h2_b, wue, up_f, SEQ, IDIM, DDIM, 0);
        act_kernel<<<2048, 256, 0, stream>>>(gate_f, up_f, topi, aff, e, act_b);
        gemm_bf16_kernel<<<dim3(DDIM / BN, SEQ / BM), 256, 0, stream>>>(act_b, wde, moe_f, SEQ, DDIM, IDIM, e ? 1 : 0);
    }

    // 10) final residual
    add_kernel<<<4096, 256, 0, stream>>>(x1_f, moe_f, out, SD);
}
